// Model_69793218560060
// MI455X (gfx1250) — matrix operand order repaired and checked by reading, not compiled, not run on hardware
//
#include <hip/hip_runtime.h>

// ---------------- Problem constants ----------------
#define BB      256          // batch
#define HH      512          // hidden
#define FH      2048         // 4*H
#define TW      192          // SEQ_LEN + PRED_LEN
#define SEQ     96
#define CIN     9            // ENC_IN + LAG
#define INW     8            // IN_SIZE
#define COUT    9
#define XROW    (SEQ * CIN)  // 864 : batch-row stride inside x_enc / y_enc

// LDS staging: 2 gates x 16 rows; row = 512 bf16 = 256 dwords + 4 pad dwords
// -> row stride 260 dwords: 16B-aligned fragments (ds_load_b128), 4-bank skew.
#define ROW_DW       260
#define STAGE_DW     (32 * ROW_DW)          // 8320 dwords
#define STAGE_BYTES  (STAGE_DW * 4)         // 33,280 B ; double buffer = 66,560 B

typedef __attribute__((ext_vector_type(16))) __bf16 v16bf;
typedef __attribute__((ext_vector_type(8)))  float  v8f;
typedef unsigned int u32x4 __attribute__((ext_vector_type(4)));
typedef int          i32x4 __attribute__((ext_vector_type(4)));
typedef int          i32x8 __attribute__((ext_vector_type(8)));

#if defined(__has_builtin)
#  if __has_builtin(__builtin_amdgcn_tensor_load_to_lds)
#    define HAVE_TDM 1
#  endif
#endif

__device__ __forceinline__ unsigned short f2bf(float f) {
    unsigned int u = __float_as_uint(f);
    u += 0x7fffu + ((u >> 16) & 1u);        // round-to-nearest-even
    return (unsigned short)(u >> 16);
}
__device__ __forceinline__ float sigm(float z) { return 1.f / (1.f + __expf(-z)); }

// ---------------------------------------------------------------------------
// TDM: stage a (2 gates x 16 rows x 512 cols) bf16 weight tile into LDS.
// 3D descriptor: x=512 elems (2B), y=16 rows stride 512, z=2 gates stride 512*512.
// pad: 4 dwords after every 256 dwords  ->  LDS row stride = 260 dwords.
// ---------------------------------------------------------------------------
__device__ __forceinline__ void tdm_issue(unsigned lds_off, const unsigned short* gptr)
{
#ifdef HAVE_TDM
    const unsigned long long ga = (unsigned long long)gptr;
    u32x4 g0;
    g0.x = 1u;                                                  // count=1, user mode
    g0.y = lds_off;                                             // lds_addr
    g0.z = (unsigned)ga;                                        // global_addr[31:0]
    g0.w = ((unsigned)(ga >> 32) & 0x01FFFFFFu) | 0x80000000u;  // addr[56:32] | type=2
    i32x8 g1;
    g1[0] = 0x07D10000;            // data_size=2B, pad_enable, interval=256dw, amount=4dw
    g1[1] = (int)(512u << 16);     // tensor_dim0 = 512
    g1[2] = (int)(16u << 16);      // tensor_dim1 = 16
    g1[3] = (int)(512u << 16);     // tile_dim0 = 512
    g1[4] = (int)((2u << 16) | 16u); // tile_dim2=2 (gates), tile_dim1=16
    g1[5] = 512;                   // tensor_dim0_stride = 512
    g1[6] = 0;                     // tensor_dim1_stride[15:0] = 0 (262144 & 0xffff)
    g1[7] = 4;                     // tensor_dim1_stride[47:16] = 0x4  (=262144)
    i32x4 g2 = (i32x4){2, 0, 0, 0};   // tensor_dim2 = 2
    i32x4 g3 = (i32x4){0, 0, 0, 0};
#  if __clang_major__ >= 23
    i32x8 g4 = (i32x8){0, 0, 0, 0, 0, 0, 0, 0};
    __builtin_amdgcn_tensor_load_to_lds(g0, g1, g2, g3, g4, 0);
#  else
    __builtin_amdgcn_tensor_load_to_lds(g0, g1, g2, g3, 0);
#  endif
#else
    (void)lds_off; (void)gptr;
#endif
}

template <int N>
__device__ __forceinline__ void tdm_wait()
{
#ifdef HAVE_TDM
#  if __has_builtin(__builtin_amdgcn_s_wait_tensorcnt)
    __builtin_amdgcn_s_wait_tensorcnt(N);          // immediate required
#  else
    asm volatile("s_wait_tensorcnt %0" :: "i"(N) : "memory");
#  endif
#endif
}

// Fallback (no TDM builtin): cooperative dword copy with the same padded layout.
__device__ __forceinline__ void copy_stage(unsigned int* sm, const unsigned int* Wdw,
                                           int grow_base)
{
#ifndef HAVE_TDM
    for (int d = threadIdx.x; d < 32 * 256; d += 256) {
        const int lrow = d >> 8, col = d & 255;                    // lrow = lg*16+n
        const int grow = grow_base + ((lrow >> 4) << 9) + (lrow & 15);
        sm[lrow * ROW_DW + col] = Wdw[grow * 256 + col];
    }
#else
    (void)sm; (void)Wdw; (void)grow_base;
#endif
}

// issue stage for phase -> buffer (wave0 only on TDM path; all threads on fallback)
#define STAGE_ISSUE(bufp, W, growbase)                                          \
    do {                                                                        \
        if (threadIdx.x < 32)                                                   \
            tdm_issue((unsigned)(unsigned long long)(bufp),                     \
                      (W) + (size_t)(growbase) * HH);                           \
        copy_stage((bufp), (const unsigned int*)(W), (growbase));               \
    } while (0)

#define STAGE_WAIT(n)                                                           \
    do { if (threadIdx.x < 32) tdm_wait<(n)>(); __syncthreads(); } while (0)

// ---------------------------------------------------------------------------
// 2-gate GEMM: acc{0,1} += A(16x512 of hsrc) * B_lds(512x16 per gate)
// A: bf16 row-major in global, ld=512.  B: LDS, padded row stride 260 dwords.
// A and B fragments use contiguous k = 16*(lane/16) + s (not the ISA table order).
// ---------------------------------------------------------------------------
__device__ __forceinline__ void gemm_tile2(v8f* acc0, v8f* acc1,
                                           const unsigned int* __restrict__ hsrc,
                                           const unsigned int* __restrict__ sm,
                                           int mrow, int lane)
{
    const int mA  = lane & 15;
    const int khA = (lane >> 4) << 4;   // 0 / 16
    const int nB  = lane & 15;
    const int khBd = (lane >> 4) << 3;  // B K-half offset in dwords (0 / 8)
    const int rowA = (mrow + mA) * HH;

    #pragma unroll 4
    for (int kk = 0; kk < 16; ++kk) {
        const int kb = kk << 5;
        union { unsigned int u[8]; v16bf v; } a;
        const int a0 = (rowA + kb + khA) >> 1;
        const int a1 = (rowA + kb + 8 + khA) >> 1;
        #pragma unroll
        for (int v2 = 0; v2 < 4; ++v2) {
            a.u[v2]     = hsrc[a0 + v2];
            a.u[4 + v2] = hsrc[a1 + v2];
        }
        if (kk < 15) __builtin_prefetch(&hsrc[a0 + 16], 0, 0);

        const int kdw = (kk << 4) + khBd;               // (kb + khB)/2, multiple of 8
        union { unsigned int u[8]; v16bf v; } b0, b1;
        const int r0 = nB * ROW_DW + kdw;               // 16B-aligned (260*n+8m)
        const int r1 = (16 + nB) * ROW_DW + kdw;
        #pragma unroll
        for (int v2 = 0; v2 < 8; ++v2) { b0.u[v2] = sm[r0 + v2]; b1.u[v2] = sm[r1 + v2]; }

        *acc0 = __builtin_amdgcn_wmma_f32_16x16x32_bf16(
            false, a.v, false, b0.v, (short)0, *acc0, false, false);
        *acc1 = __builtin_amdgcn_wmma_f32_16x16x32_bf16(
            false, a.v, false, b1.v, (short)0, *acc1, false, false);
    }
}

// ---------------------------------------------------------------------------
// One-time: f32->bf16 weight conversion, bias fusion, state zeroing
// ---------------------------------------------------------------------------
__global__ void init_kernel(const float* __restrict__ Whh0, const float* __restrict__ Wih1,
                            const float* __restrict__ Whh1,
                            const float* __restrict__ bih0, const float* __restrict__ bhh0,
                            const float* __restrict__ bih1, const float* __restrict__ bhh1,
                            unsigned short* Whh0b, unsigned short* Wih1b, unsigned short* Whh1b,
                            float* bs0, float* bs1,
                            unsigned short* h0b0, unsigned short* h1b0,
                            float* c0, float* c1)
{
    int i = blockIdx.x * blockDim.x + threadIdx.x;
    if (i < FH * HH) {
        Whh0b[i] = f2bf(Whh0[i]);
        Wih1b[i] = f2bf(Wih1[i]);
        Whh1b[i] = f2bf(Whh1[i]);
    }
    if (i < FH) { bs0[i] = bih0[i] + bhh0[i]; bs1[i] = bih1[i] + bhh1[i]; }
    if (i < BB * HH) { h0b0[i] = 0; h1b0[i] = 0; c0[i] = 0.f; c1[i] = 0.f; }
}

// ---------------------------------------------------------------------------
// Layer 0 cell.  grid = 64 blocks x 256 thr.  block = (m-half, n_tile):
// all 8 waves share one n_tile; B staged by TDM, double-buffered + pipelined.
// ---------------------------------------------------------------------------
__global__ __launch_bounds__(256, 2)
void lstm_l0(const float* __restrict__ x_t,
             const unsigned short* __restrict__ Whh_bf,   // (2048x512) bf16
             const float* __restrict__ bs,                // bih+bhh (2048)
             const float* __restrict__ Wih0,              // (2048x8) f32
             const unsigned int* __restrict__ hprev_bf,
             unsigned short* __restrict__ hnew_bf,
             float* __restrict__ hf, float* __restrict__ c)
{
    extern __shared__ __align__(16) unsigned int smem[];  // 2 * STAGE_DW
    unsigned int* buf0 = smem;
    unsigned int* buf1 = smem + STAGE_DW;

    const int lane   = threadIdx.x & 31;
    const int n_tile = blockIdx.x & 31;
    const int mrow   = (((blockIdx.x >> 5) << 3) + (threadIdx.x >> 5)) << 4;
    const int ncol0  = n_tile << 4;

    v8f acc[4] = {v8f{0.f}, v8f{0.f}, v8f{0.f}, v8f{0.f}};

    STAGE_ISSUE(buf0, Whh_bf, ncol0);              // phase 0: gates 0,1
    STAGE_ISSUE(buf1, Whh_bf, 2 * HH + ncol0);     // phase 1: gates 2,3 (in flight)
    STAGE_WAIT(1);                                 // phase 0 landed
    gemm_tile2(&acc[0], &acc[1], hprev_bf, buf0, mrow, lane);
    STAGE_WAIT(0);                                 // phase 1 landed
    gemm_tile2(&acc[2], &acc[3], hprev_bf, buf1, mrow, lane);

    const int nh = ncol0 + (lane & 15);
    const float bsI = bs[nh], bsF = bs[HH + nh], bsG = bs[2 * HH + nh], bsO = bs[3 * HH + nh];
    float wI[INW], wF[INW], wG[INW], wO[INW];
    #pragma unroll
    for (int k = 0; k < INW; ++k) {
        wI[k] = Wih0[nh * INW + k];
        wF[k] = Wih0[(HH + nh) * INW + k];
        wG[k] = Wih0[(2 * HH + nh) * INW + k];
        wO[k] = Wih0[(3 * HH + nh) * INW + k];
    }
    const int mhi = (lane >> 4) << 3;
    #pragma unroll
    for (int r = 0; r < 8; ++r) {
        const int b = mrow + mhi + r;
        const float* xr = x_t + b * XROW;
        float gi = acc[0][r] + bsI, gf = acc[1][r] + bsF;
        float gg = acc[2][r] + bsG, go = acc[3][r] + bsO;
        #pragma unroll
        for (int k = 0; k < INW; ++k) {
            const float xv = xr[k];
            gi = fmaf(wI[k], xv, gi); gf = fmaf(wF[k], xv, gf);
            gg = fmaf(wG[k], xv, gg); go = fmaf(wO[k], xv, go);
        }
        const int idx = b * HH + nh;
        const float cn = sigm(gf) * c[idx] + sigm(gi) * tanhf(gg);
        const float hv = sigm(go) * tanhf(cn);
        c[idx]  = cn;
        hf[idx] = hv;
        hnew_bf[idx] = f2bf(hv);
    }
}

// ---------------------------------------------------------------------------
// Layer 1 cell: two K=512 GEMMs; 4 pipelined TDM phases over 2 LDS buffers.
// ---------------------------------------------------------------------------
__global__ __launch_bounds__(256, 2)
void lstm_l1(const unsigned int* __restrict__ h0cur_bf,
             const unsigned short* __restrict__ Wih_bf,
             const unsigned int* __restrict__ h1prev_bf,
             const unsigned short* __restrict__ Whh_bf,
             const float* __restrict__ bs,
             unsigned short* __restrict__ hnew_bf,
             float* __restrict__ hf, float* __restrict__ c)
{
    extern __shared__ __align__(16) unsigned int smem[];  // 2 * STAGE_DW
    unsigned int* buf0 = smem;
    unsigned int* buf1 = smem + STAGE_DW;

    const int lane   = threadIdx.x & 31;
    const int n_tile = blockIdx.x & 31;
    const int mrow   = (((blockIdx.x >> 5) << 3) + (threadIdx.x >> 5)) << 4;
    const int ncol0  = n_tile << 4;

    v8f acc[4] = {v8f{0.f}, v8f{0.f}, v8f{0.f}, v8f{0.f}};

    STAGE_ISSUE(buf0, Wih_bf, ncol0);                  // phase 0
    STAGE_ISSUE(buf1, Wih_bf, 2 * HH + ncol0);         // phase 1
    STAGE_WAIT(1);                                     // phase 0 landed
    gemm_tile2(&acc[0], &acc[1], h0cur_bf, buf0, mrow, lane);
    __syncthreads();                                   // buf0 consumed
    STAGE_ISSUE(buf0, Whh_bf, ncol0);                  // phase 2
    STAGE_WAIT(1);                                     // phase 1 landed
    gemm_tile2(&acc[2], &acc[3], h0cur_bf, buf1, mrow, lane);
    __syncthreads();                                   // buf1 consumed
    STAGE_ISSUE(buf1, Whh_bf, 2 * HH + ncol0);         // phase 3
    STAGE_WAIT(1);                                     // phase 2 landed
    gemm_tile2(&acc[0], &acc[1], h1prev_bf, buf0, mrow, lane);
    STAGE_WAIT(0);                                     // phase 3 landed
    gemm_tile2(&acc[2], &acc[3], h1prev_bf, buf1, mrow, lane);

    const int nh = ncol0 + (lane & 15);
    const float bsI = bs[nh], bsF = bs[HH + nh], bsG = bs[2 * HH + nh], bsO = bs[3 * HH + nh];
    const int mhi = (lane >> 4) << 3;
    #pragma unroll
    for (int r = 0; r < 8; ++r) {
        const int b = mrow + mhi + r;
        const int idx = b * HH + nh;
        const float gi = acc[0][r] + bsI, gf = acc[1][r] + bsF;
        const float gg = acc[2][r] + bsG, go = acc[3][r] + bsO;
        const float cn = sigm(gf) * c[idx] + sigm(gi) * tanhf(gg);
        const float hv = sigm(go) * tanhf(cn);
        c[idx]  = cn;
        hf[idx] = hv;
        hnew_bf[idx] = f2bf(hv);
    }
}

// ---------------------------------------------------------------------------
// Per-step projection: out[b,t,co] = bout[co] + sum_j h0*W[co,2j] + h1*W[co,2j+1]
// ---------------------------------------------------------------------------
__global__ void proj_kernel(const float* __restrict__ hf0, const float* __restrict__ hf1,
                            const float* __restrict__ Wout, const float* __restrict__ bout,
                            float* __restrict__ out, int t)
{
    const int idx = blockIdx.x * blockDim.x + threadIdx.x;
    if (idx >= BB * COUT) return;
    const int b = idx / COUT, co = idx % COUT;
    const float* w  = Wout + co * (2 * HH);
    const float* p0 = hf0 + b * HH;
    const float* p1 = hf1 + b * HH;
    float s = bout[co];
    #pragma unroll 4
    for (int j = 0; j < HH; ++j)
        s = fmaf(p0[j], w[2 * j], fmaf(p1[j], w[2 * j + 1], s));
    out[(b * TW + t) * COUT + co] = s;
}

// ---------------------------------------------------------------------------
// Second output: batch = concat(x_enc, y_enc) -> d_out[B*TW*9 ...]
// ---------------------------------------------------------------------------
__global__ void batch_copy(const float* __restrict__ x_enc, const float* __restrict__ y_enc,
                           float* __restrict__ out2)
{
    const int idx = blockIdx.x * blockDim.x + threadIdx.x;
    if (idx >= BB * TW * CIN) return;
    const int b = idx / (TW * CIN);
    const int rem = idx % (TW * CIN);
    const int t = rem / CIN, k = rem % CIN;
    out2[idx] = (t < SEQ) ? x_enc[(b * SEQ + t) * CIN + k]
                          : y_enc[(b * SEQ + (t - SEQ)) * CIN + k];
}

// ---------------------------------------------------------------------------
extern "C" void kernel_launch(void* const* d_in, const int* in_sizes, int n_in,
                              void* d_out, int out_size, void* d_ws, size_t ws_size,
                              hipStream_t stream)
{
    const float* x_enc = (const float*)d_in[0];
    const float* y_enc = (const float*)d_in[3];
    const float* Wih0  = (const float*)d_in[5];
    const float* Whh0  = (const float*)d_in[6];
    const float* bih0  = (const float*)d_in[7];
    const float* bhh0  = (const float*)d_in[8];
    const float* Wih1  = (const float*)d_in[9];
    const float* Whh1  = (const float*)d_in[10];
    const float* bih1  = (const float*)d_in[11];
    const float* bhh1  = (const float*)d_in[12];
    const float* Wout  = (const float*)d_in[13];
    const float* bout  = (const float*)d_in[14];
    float* out = (float*)d_out;

    // ---- workspace layout (bytes) ----
    char* ws = (char*)d_ws;
    const size_t szW   = (size_t)FH * HH * sizeof(unsigned short);   // 2 MB each
    const size_t szBs  = (size_t)FH * sizeof(float);
    const size_t szHbf = (size_t)BB * HH * sizeof(unsigned short);   // one buffer
    const size_t szF   = (size_t)BB * HH * sizeof(float);

    unsigned short* Whh0b = (unsigned short*)(ws);
    unsigned short* Wih1b = (unsigned short*)(ws + szW);
    unsigned short* Whh1b = (unsigned short*)(ws + 2 * szW);
    float* bs0 = (float*)(ws + 3 * szW);
    float* bs1 = (float*)(ws + 3 * szW + szBs);
    unsigned short* h0bf = (unsigned short*)(ws + 3 * szW + 2 * szBs);              // x2
    unsigned short* h1bf = (unsigned short*)(ws + 3 * szW + 2 * szBs + 2 * szHbf);  // x2
    float* h0f = (float*)(ws + 3 * szW + 2 * szBs + 4 * szHbf);
    float* h1f = (float*)((char*)h0f + szF);
    float* c0  = (float*)((char*)h1f + szF);
    float* c1  = (float*)((char*)c0 + szF);

    init_kernel<<<(FH * HH + 255) / 256, 256, 0, stream>>>(
        Whh0, Wih1, Whh1, bih0, bhh0, bih1, bhh1,
        Whh0b, Wih1b, Whh1b, bs0, bs1, h0bf, h1bf, c0, c1);

    const size_t shmem = 2 * STAGE_BYTES;   // 66,560 B dynamic LDS (hw: 320 KB/WGP)

    for (int t = 0; t < TW; ++t) {
        const int rd = t & 1, wr = rd ^ 1;
        const float* x_t = (t < SEQ) ? (x_enc + t * CIN) : (y_enc + (t - SEQ) * CIN);

        lstm_l0<<<64, 256, shmem, stream>>>(
            x_t, Whh0b, bs0, Wih0,
            (const unsigned int*)(h0bf + rd * (BB * HH)),
            h0bf + wr * (BB * HH), h0f, c0);

        lstm_l1<<<64, 256, shmem, stream>>>(
            (const unsigned int*)(h0bf + wr * (BB * HH)),
            Wih1b,
            (const unsigned int*)(h1bf + rd * (BB * HH)),
            Whh1b,
            bs1,
            h1bf + wr * (BB * HH), h1f, c1);

        proj_kernel<<<(BB * COUT + 255) / 256, 256, 0, stream>>>(h0f, h1f, Wout, bout, out, t);
    }

    batch_copy<<<(BB * TW * CIN + 255) / 256, 256, 0, stream>>>(
        x_enc, y_enc, out + (size_t)BB * TW * COUT);
}
